// Mamba2Block_25563645346551
// MI455X (gfx1250) — matrix operand order repaired and checked by reading, not compiled, not run on hardware
//
#include <hip/hip_runtime.h>
#include <hip/hip_bf16.h>

// ---------------------------------------------------------------------------
// Mamba2 bidirectional block for gfx1250 (CDNA5, wave32, WMMA + async-LDS)
// ---------------------------------------------------------------------------

#define D_MODEL 768
#define D_STATE 64
#define D_CONV  4
#define D_INNER 1536
#define BATCH   2
#define SEQ     1024
#define NTOK    (BATCH * SEQ)               // 2048
#define X_PROJ_OUT (2 * D_STATE + D_INNER)  // 1664 (only first 128 cols used)

typedef __attribute__((ext_vector_type(16))) _Float16 v16h;
typedef __attribute__((ext_vector_type(8)))  _Float16 v8h;
typedef __attribute__((ext_vector_type(8)))  float    v8f;
typedef __attribute__((ext_vector_type(4)))  int      v4i;

#if defined(__has_builtin)
#if __has_builtin(__builtin_amdgcn_global_load_async_to_lds_b128) && \
    __has_builtin(__builtin_amdgcn_s_wait_asynccnt)
#define USE_ASYNC_A 1
#endif
#endif
#ifndef USE_ASYNC_A
#define USE_ASYNC_A 0
#endif

#if USE_ASYNC_A
// builtin expects int4 pointers in AS1 (global) / AS3 (LDS).
// Go through uintptr_t: generic->AS1 is bit-identical; generic LDS pointers
// are {aperture_hi | offset_lo}, so 32-bit truncation yields the AS3 offset
// (same lowering as addrspacecast generic->local).
typedef __attribute__((address_space(1))) v4i* gas_v4i_p;
typedef __attribute__((address_space(3))) v4i* las_v4i_p;
#define ASYNC_CP_B128(gsrc, ldst)                                          \
    __builtin_amdgcn_global_load_async_to_lds_b128(                        \
        (gas_v4i_p)(uintptr_t)(gsrc), (las_v4i_p)(uintptr_t)(ldst), 0, 0)
#endif

// ------------------------------ helpers ------------------------------------

__device__ __forceinline__ float silu_f(float v) {
    // v / (1 + e^-v) via fast v_rcp_f32
    return v * __builtin_amdgcn_rcpf(1.0f + __expf(-v));
}

__device__ __forceinline__ float softplus_f(float t) {
    return (t > 20.0f) ? t : __logf(1.0f + __expf(t));
}

// f32 -> f16 convert (weight staging)
__global__ void cvt_f32_to_f16_kernel(const float* __restrict__ src,
                                      _Float16* __restrict__ dst, int n) {
    int i = blockIdx.x * blockDim.x + threadIdx.x;
    if (i < n) dst[i] = (_Float16)src[i];
}

// gather first `cols` columns of a row-major [rows, srcStride] f32 matrix
__global__ void gather_cols_f16_kernel(const float* __restrict__ src,
                                       _Float16* __restrict__ dst,
                                       int rows, int srcStride, int cols) {
    int i = blockIdx.x * blockDim.x + threadIdx.x;
    int n = rows * cols;
    if (i >= n) return;
    int r = i / cols, c = i - r * cols;
    dst[i] = (_Float16)src[(size_t)r * srcStride + c];
}

// ------------------------------ layernorm ----------------------------------

__global__ void __launch_bounds__(256)
ln_kernel(const float* __restrict__ x, const float* __restrict__ w,
          const float* __restrict__ b, _Float16* __restrict__ xn) {
    __shared__ float red[256];
    const int row = blockIdx.x;            // token index, 0..2047
    const int tid = threadIdx.x;
    const float* xr = x + (size_t)row * D_MODEL;

    float s = 0.f, ss = 0.f;
    for (int c = tid; c < D_MODEL; c += 256) {
        float v = xr[c];
        s += v; ss += v * v;
    }
    red[tid] = s; __syncthreads();
    for (int o = 128; o > 0; o >>= 1) {
        if (tid < o) red[tid] += red[tid + o];
        __syncthreads();
    }
    float mean = red[0] / (float)D_MODEL;
    __syncthreads();
    red[tid] = ss; __syncthreads();
    for (int o = 128; o > 0; o >>= 1) {
        if (tid < o) red[tid] += red[tid + o];
        __syncthreads();
    }
    float var = red[0] / (float)D_MODEL - mean * mean;
    float rs  = rsqrtf(var + 1e-5f);
    for (int c = tid; c < D_MODEL; c += 256) {
        float v = (xr[c] - mean) * rs * w[c] + b[c];
        xn[(size_t)row * D_MODEL + c] = (_Float16)v;
    }
}

// ---------------------- WMMA f16->f32 tiled GEMM ---------------------------
// C[M,N] = A[M,K] * B[K,N]; row-major f16 in; f32 accumulate.
// Block tile 128x64, K-step 32, 256 threads = 8 waves (4x2),
// each wave 32x32 = 2x2 v_wmma_f32_16x16x32_f16 accumulators.
// B tile stored TRANSPOSED in LDS so both fragments are ds_load_b128 pairs.
// A tile staged via async global->LDS (ping-pong) when available.

#define BLK_M 128
#define BLK_N 64
#define BLK_K 32
#define LDS_KP (BLK_K + 8)   // 40 halfs = 80B rows (16B-aligned)

__device__ __forceinline__ void wmma_compute_step(
    const _Float16 (*__restrict__ Asl)[LDS_KP],
    const _Float16 (*__restrict__ Btl)[LDS_KP],
    int wm, int wn, int l15, int lhi, v8f acc[2][2]) {
    union Frag { v16h v; v8h h[2]; };
    Frag af[2], bf[2];
#pragma unroll
    for (int ni = 0; ni < 2; ++ni) {
        int col = wn * 32 + ni * 16 + l15;
        bf[ni].h[0] = *(const v8h*)(&Btl[col][lhi * 8]);
        bf[ni].h[1] = *(const v8h*)(&Btl[col][lhi * 8 + 16]);
    }
#pragma unroll
    for (int mi = 0; mi < 2; ++mi) {
        int row = wm * 32 + mi * 16 + l15;
        af[mi].h[0] = *(const v8h*)(&Asl[row][lhi * 8]);
        af[mi].h[1] = *(const v8h*)(&Asl[row][lhi * 8 + 16]);
    }
#pragma unroll
    for (int mi = 0; mi < 2; ++mi)
#pragma unroll
        for (int ni = 0; ni < 2; ++ni)
            acc[mi][ni] = __builtin_amdgcn_wmma_f32_16x16x32_f16(
                false, af[mi].v, false, bf[ni].v, (short)0, acc[mi][ni],
                false, false);
}

// MODE 0: in_proj split  (col<1536 -> out0 x_in f32 ; else out1 = silu -> zs)
// MODE 1: ssm split      (col<64   -> out16 dpre f16; else out0 Bm f32)
// MODE 2: delta          (out0 = softplus(v + bias[col]))
// MODE 3: out_proj       (out0 = v + resid)
template <int MODE>
__global__ void __launch_bounds__(256)
gemm_wmma_kernel(const _Float16* __restrict__ A, const _Float16* __restrict__ B,
                 int M, int N, int K,
                 float* __restrict__ out0, float* __restrict__ out1,
                 const float* __restrict__ bias, const float* __restrict__ resid,
                 _Float16* __restrict__ out16) {
    const int tid  = threadIdx.x;
    const int lane = tid & 31;
    const int wave = tid >> 5;       // 0..7
    const int wm   = wave >> 1;      // 0..3 : 32-row band
    const int wn   = wave & 1;       // 0..1 : 32-col band
    const int l15  = lane & 15;
    const int lhi  = lane >> 4;      // 0 | 1
    const int blockM = blockIdx.y * BLK_M;
    const int blockN = blockIdx.x * BLK_N;

    // global staging slots: A 16 halfs/thread, B 8 halfs/thread
    const int ea = tid * 16, ra = ea >> 5, ca = ea & 31;   // A: 128x32
    const int eb = tid * 8,  rb = eb >> 6, cb = eb & 63;   // B: 32x64

    v8f acc[2][2] = {};
    const int nIter = K / BLK_K;

#if USE_ASYNC_A
    __shared__ __align__(16) _Float16 As[2][BLK_M][LDS_KP];
    __shared__ __align__(16) _Float16 Bt[2][BLK_N][LDS_KP];

    v8h bReg = *(const v8h*)(&B[(size_t)rb * N + blockN + cb]);
    ASYNC_CP_B128(&A[(size_t)(blockM + ra) * K + ca],     &As[0][ra][ca]);
    ASYNC_CP_B128(&A[(size_t)(blockM + ra) * K + ca + 8], &As[0][ra][ca + 8]);

    for (int i = 0; i < nIter; ++i) {
        const int cur = i & 1;
        const int k0  = i * BLK_K;
        // transpose-store B tile (safe: Bt[cur] last read 2 barriers ago)
#pragma unroll
        for (int j = 0; j < 8; ++j) Bt[cur][cb + j][rb] = bReg[j];
        __builtin_amdgcn_s_wait_asynccnt(0);   // own A-tile async copies done
        __syncthreads();                       // everyone's tile visible
        if (i + 1 < nIter) {
            const int kn = k0 + BLK_K;
            bReg = *(const v8h*)(&B[(size_t)(kn + rb) * N + blockN + cb]);
            ASYNC_CP_B128(&A[(size_t)(blockM + ra) * K + kn + ca],
                          &As[cur ^ 1][ra][ca]);
            ASYNC_CP_B128(&A[(size_t)(blockM + ra) * K + kn + ca + 8],
                          &As[cur ^ 1][ra][ca + 8]);
            if (kn + BLK_K < K) {
                __builtin_prefetch(&A[(size_t)(blockM + ra) * K + kn + BLK_K + ca], 0, 3);
                __builtin_prefetch(&B[(size_t)(kn + BLK_K + rb) * N + blockN + cb], 0, 3);
            }
        }
        wmma_compute_step(As[cur], Bt[cur], wm, wn, l15, lhi, acc);
    }
#else
    __shared__ __align__(16) _Float16 As1[BLK_M][LDS_KP];
    __shared__ __align__(16) _Float16 Bt1[BLK_N][LDS_KP];

    // register-staged pipeline: global loads for tile i+1 overlap compute of i
    v8h a0 = *(const v8h*)(&A[(size_t)(blockM + ra) * K + ca]);
    v8h a1 = *(const v8h*)(&A[(size_t)(blockM + ra) * K + ca + 8]);
    v8h b0 = *(const v8h*)(&B[(size_t)rb * N + blockN + cb]);

    for (int i = 0; i < nIter; ++i) {
        const int k0 = i * BLK_K;
        __syncthreads();                 // LDS free (all done with tile i-1)
        *(v8h*)(&As1[ra][ca])     = a0;
        *(v8h*)(&As1[ra][ca + 8]) = a1;
#pragma unroll
        for (int j = 0; j < 8; ++j) Bt1[cb + j][rb] = b0[j];
        __syncthreads();                 // tile visible
        if (i + 1 < nIter) {
            const int kn = k0 + BLK_K;
            a0 = *(const v8h*)(&A[(size_t)(blockM + ra) * K + kn + ca]);
            a1 = *(const v8h*)(&A[(size_t)(blockM + ra) * K + kn + ca + 8]);
            b0 = *(const v8h*)(&B[(size_t)(kn + rb) * N + blockN + cb]);
            if (kn + BLK_K < K) {
                __builtin_prefetch(&A[(size_t)(blockM + ra) * K + kn + BLK_K + ca], 0, 3);
                __builtin_prefetch(&B[(size_t)(kn + BLK_K + rb) * N + blockN + cb], 0, 3);
            }
        }
        wmma_compute_step(As1, Bt1, wm, wn, l15, lhi, acc);
    }
#endif

    // epilogue: VGPR v -> M = v + 8*lhi ; N = l15
#pragma unroll
    for (int mi = 0; mi < 2; ++mi) {
#pragma unroll
        for (int ni = 0; ni < 2; ++ni) {
#pragma unroll
            for (int v = 0; v < 8; ++v) {
                int row = blockM + wm * 32 + mi * 16 + v + lhi * 8;
                int col = blockN + wn * 32 + ni * 16 + l15;
                float val = acc[mi][ni][v];
                if (MODE == 0) {
                    if (col < D_INNER)
                        out0[(size_t)row * D_INNER + col] = val;
                    else
                        out1[(size_t)row * D_INNER + (col - D_INNER)] = silu_f(val);
                } else if (MODE == 1) {
                    if (col < D_STATE)
                        out16[(size_t)row * D_STATE + col] = (_Float16)val;
                    else
                        out0[(size_t)row * D_STATE + (col - D_STATE)] = val;
                } else if (MODE == 2) {
                    out0[(size_t)row * N + col] = softplus_f(val + bias[col]);
                } else {
                    out0[(size_t)row * N + col] =
                        val + resid[(size_t)row * N + col];
                }
            }
        }
    }
}

// ----------------------- causal depthwise conv + silu ----------------------

__global__ void __launch_bounds__(256)
conv_silu_kernel(const float* __restrict__ xin, const float* __restrict__ w,
                 const float* __restrict__ bias, float* __restrict__ xc32,
                 _Float16* __restrict__ xc16) {
    int idx = blockIdx.x * blockDim.x + threadIdx.x;
    if (idx >= BATCH * SEQ * D_INNER) return;
    int d   = idx % D_INNER;
    int bt  = idx / D_INNER;
    int t   = bt % SEQ;
    int b   = bt / SEQ;

    float acc = bias[d];
#pragma unroll
    for (int k = 0; k < D_CONV; ++k) {
        int tt = t - (D_CONV - 1) + k;
        if (tt >= 0)
            acc += w[d * D_CONV + k] *
                   xin[((size_t)(b * SEQ + tt)) * D_INNER + d];
    }
    float y = silu_f(acc);
    xc32[idx] = y;
    xc16[idx] = (_Float16)y;
}

// ------------------------------ selective scan -----------------------------
// grid (D_INNER/256, BATCH); thread owns one channel; h[64] in registers.
// B_t (== C_t) staged 16 timesteps at a time through LDS (2 barriers / 16
// steps); A[n] = -exp(A_log[n]) cached in LDS.

#define T_CHUNK 16

__global__ void __launch_bounds__(256)
scan_kernel(const float* __restrict__ xc, const float* __restrict__ delta,
            const float* __restrict__ Bm, const float* __restrict__ Dp,
            const float* __restrict__ A_log, float* __restrict__ y,
            int reverse) {
    __shared__ float sB[T_CHUNK * D_STATE];   // 4 KB
    __shared__ float sA[D_STATE];

    const int d = blockIdx.x * 256 + threadIdx.x;
    const int b = blockIdx.y;
    const float Dv = Dp[d];

    if (threadIdx.x < D_STATE)
        sA[threadIdx.x] = -__expf(A_log[threadIdx.x]);

    float h[D_STATE];
#pragma unroll
    for (int n = 0; n < D_STATE; ++n) h[n] = 0.f;

    for (int c0 = 0; c0 < SEQ; c0 += T_CHUNK) {
        const int tbase = reverse ? (SEQ - c0 - T_CHUNK) : c0;
        __syncthreads();
        {   // 16 timesteps x 64 states = 1024 floats, 4 per thread
            const int i = threadIdx.x * 4;
            *(float4*)&sB[i] =
                *(const float4*)&Bm[((size_t)(b * SEQ + tbase)) * D_STATE + i];
        }
        __syncthreads();
        for (int s = 0; s < T_CHUNK; ++s) {
            const int tl = reverse ? (T_CHUNK - 1 - s) : s;
            const size_t tok = (size_t)(b * SEQ + tbase + tl);
            const float* bRow = &sB[tl * D_STATE];
            const float xv = xc[tok * D_INNER + d];
            const float dl = delta[tok * D_INNER + d];
            const float dx = dl * xv;
            float yv = 0.f;
#pragma unroll
            for (int n = 0; n < D_STATE; ++n) {
                float bn  = bRow[n];
                float dAn = __expf(dl * sA[n]);
                h[n] = dAn * h[n] + dx * bn;
                yv  += h[n] * bn;
            }
            y[tok * D_INNER + d] = yv + Dv * xv;
        }
    }
}

// ------------------------- gate/concat combine -----------------------------

__global__ void __launch_bounds__(256)
combine_kernel(const float* __restrict__ yf, const float* __restrict__ yr,
               const float* __restrict__ zs, _Float16* __restrict__ u) {
    int idx = blockIdx.x * blockDim.x + threadIdx.x;
    if (idx >= NTOK * 2 * D_INNER) return;
    int col = idx % (2 * D_INNER);
    int row = idx / (2 * D_INNER);
    int c   = (col < D_INNER) ? col : (col - D_INNER);
    const float* ysrc = (col < D_INNER) ? yf : yr;
    float v = ysrc[(size_t)row * D_INNER + c] * zs[(size_t)row * D_INNER + c];
    u[idx] = (_Float16)v;
}

// ------------------------------- launch ------------------------------------

extern "C" void kernel_launch(void* const* d_in, const int* in_sizes, int n_in,
                              void* d_out, int out_size, void* d_ws, size_t ws_size,
                              hipStream_t stream) {
    const float* x          = (const float*)d_in[0];
    const float* ln_w       = (const float*)d_in[1];
    const float* ln_b       = (const float*)d_in[2];
    const float* in_proj_w  = (const float*)d_in[3];
    const float* conv_w     = (const float*)d_in[4];
    const float* conv_b     = (const float*)d_in[5];
    const float* conv_rev_w = (const float*)d_in[6];
    const float* conv_rev_b = (const float*)d_in[7];
    const float* x_proj_w   = (const float*)d_in[8];
    const float* x_proj_rev = (const float*)d_in[9];
    const float* dt_w       = (const float*)d_in[10];
    const float* dt_b       = (const float*)d_in[11];
    const float* A_log      = (const float*)d_in[12];
    const float* Dp         = (const float*)d_in[13];
    const float* out_proj_w = (const float*)d_in[14];
    float* out = (float*)d_out;

    char* ws = (char*)d_ws;
    size_t off = 0;
    auto take = [&](size_t bytes) -> void* {
        off = (off + 255) & ~(size_t)255;
        void* p = ws + off;
        off += bytes;
        return p;
    };

    _Float16* wInP16  = (_Float16*)take((size_t)D_MODEL * 2 * D_INNER * 2);
    _Float16* wXpF16  = (_Float16*)take((size_t)D_INNER * 128 * 2);
    _Float16* wXpR16  = (_Float16*)take((size_t)D_INNER * 128 * 2);
    _Float16* wDt16   = (_Float16*)take((size_t)D_STATE * D_INNER * 2);
    _Float16* wOut16  = (_Float16*)take((size_t)2 * D_INNER * D_MODEL * 2);
    _Float16* xn16    = (_Float16*)take((size_t)NTOK * D_MODEL * 2);
    float*    xin32   = (float*)take((size_t)NTOK * D_INNER * 4);
    float*    zs32    = (float*)take((size_t)NTOK * D_INNER * 4);
    float*    xc32    = (float*)take((size_t)NTOK * D_INNER * 4);
    _Float16* xc16    = (_Float16*)take((size_t)NTOK * D_INNER * 2);
    _Float16* dpre16  = (_Float16*)take((size_t)NTOK * D_STATE * 2);
    float*    Bm32    = (float*)take((size_t)NTOK * D_STATE * 4);
    float*    delta32 = (float*)take((size_t)NTOK * D_INNER * 4);
    float*    yf32    = (float*)take((size_t)NTOK * D_INNER * 4);
    float*    yr32    = (float*)take((size_t)NTOK * D_INNER * 4);
    _Float16* u16     = (_Float16*)take((size_t)NTOK * 2 * D_INNER * 2);

    // --- stage weights to f16 ---
    {
        int n = D_MODEL * 2 * D_INNER;     // 2359296
        cvt_f32_to_f16_kernel<<<(n + 255) / 256, 256, 0, stream>>>(in_proj_w, wInP16, n);
        n = D_STATE * D_INNER;             // 98304
        cvt_f32_to_f16_kernel<<<(n + 255) / 256, 256, 0, stream>>>(dt_w, wDt16, n);
        n = 2 * D_INNER * D_MODEL;         // 2359296
        cvt_f32_to_f16_kernel<<<(n + 255) / 256, 256, 0, stream>>>(out_proj_w, wOut16, n);
        n = D_INNER * 128;                 // 196608
        gather_cols_f16_kernel<<<(n + 255) / 256, 256, 0, stream>>>(x_proj_w,   wXpF16, D_INNER, X_PROJ_OUT, 128);
        gather_cols_f16_kernel<<<(n + 255) / 256, 256, 0, stream>>>(x_proj_rev, wXpR16, D_INNER, X_PROJ_OUT, 128);
    }

    // --- layernorm ---
    ln_kernel<<<NTOK, 256, 0, stream>>>(x, ln_w, ln_b, xn16);

    // --- in_proj GEMM: [2048,768] x [768,3072] -> split x_in / silu(z) ---
    gemm_wmma_kernel<0><<<dim3(2 * D_INNER / BLK_N, NTOK / BLK_M), 256, 0, stream>>>(
        xn16, wInP16, NTOK, 2 * D_INNER, D_MODEL, xin32, zs32, nullptr, nullptr, nullptr);

    const int conv_blocks = (NTOK * D_INNER + 255) / 256;

    // ===================== forward direction =====================
    conv_silu_kernel<<<conv_blocks, 256, 0, stream>>>(xin32, conv_w, conv_b, xc32, xc16);
    gemm_wmma_kernel<1><<<dim3(128 / BLK_N, NTOK / BLK_M), 256, 0, stream>>>(
        xc16, wXpF16, NTOK, 128, D_INNER, Bm32, nullptr, nullptr, nullptr, dpre16);
    gemm_wmma_kernel<2><<<dim3(D_INNER / BLK_N, NTOK / BLK_M), 256, 0, stream>>>(
        dpre16, wDt16, NTOK, D_INNER, D_STATE, delta32, nullptr, dt_b, nullptr, nullptr);
    scan_kernel<<<dim3(D_INNER / 256, BATCH), 256, 0, stream>>>(
        xc32, delta32, Bm32, Dp, A_log, yf32, 0);

    // ===================== reverse direction =====================
    conv_silu_kernel<<<conv_blocks, 256, 0, stream>>>(xin32, conv_rev_w, conv_rev_b, xc32, xc16);
    gemm_wmma_kernel<1><<<dim3(128 / BLK_N, NTOK / BLK_M), 256, 0, stream>>>(
        xc16, wXpR16, NTOK, 128, D_INNER, Bm32, nullptr, nullptr, nullptr, dpre16);
    gemm_wmma_kernel<2><<<dim3(D_INNER / BLK_N, NTOK / BLK_M), 256, 0, stream>>>(
        dpre16, wDt16, NTOK, D_INNER, D_STATE, delta32, nullptr, dt_b, nullptr, nullptr);
    scan_kernel<<<dim3(D_INNER / 256, BATCH), 256, 0, stream>>>(
        xc32, delta32, Bm32, Dp, A_log, yr32, 1);

    // --- gate + concat ---
    {
        int n = NTOK * 2 * D_INNER;
        combine_kernel<<<(n + 255) / 256, 256, 0, stream>>>(yf32, yr32, zs32, u16);
    }

    // --- out_proj GEMM + residual: [2048,3072] x [3072,768] + x ---
    gemm_wmma_kernel<3><<<dim3(D_MODEL / BLK_N, NTOK / BLK_M), 256, 0, stream>>>(
        u16, wOut16, NTOK, D_MODEL, 2 * D_INNER, out, nullptr, nullptr, x, nullptr);
}
